// RaftLarge_48919677501863
// MI455X (gfx1250) — matrix operand order repaired and checked by reading, not compiled, not run on hardware
//
#include <hip/hip_runtime.h>
#include <cstdint>
#include <cstddef>

// ---------------------------------------------------------------- types
typedef __attribute__((ext_vector_type(16))) _Float16 v16h;
typedef __attribute__((ext_vector_type(8)))  _Float16 v8h;
typedef __attribute__((ext_vector_type(8)))  float    v8f;
typedef __attribute__((ext_vector_type(4)))  unsigned int v4u;
typedef __attribute__((ext_vector_type(8)))  int      v8i;
typedef __attribute__((ext_vector_type(4)))  int      v4i;

#define BN 2
#define CDIV(a,b) (((a)+(b)-1)/(b))
#define LDSB 48   // LDS B-tile row stride in halves (96B: 32B-aligned, bank-skewed)

// ---------------------------------------------------------------- device helpers
__device__ __forceinline__ float act_f(float v, int act) {
  if (act == 1) return v > 0.f ? v : 0.f;
  if (act == 2) return 1.f / (1.f + __expf(-v));
  if (act == 3) return tanhf(v);
  return v;
}

// ---------------------------------------------------------------- TDM weight prefetch (async tensor path)
__global__ void tdm_prefetch_kernel(const float* src, float* sink) {
#if __has_builtin(__builtin_amdgcn_tensor_load_to_lds)
  __shared__ alignas(16) float lbuf[64];
  unsigned long long ga = (unsigned long long)(uintptr_t)src;
  v4u g0;
  g0[0] = 1u;                                   // count=1 (valid descriptor)
  g0[1] = 0u;                                   // lds_addr = 0
  g0[2] = (unsigned)(ga & 0xffffffffu);         // global_addr[31:0]
  g0[3] = ((unsigned)((ga >> 32) & 0x01ffffffu)) | 0x80000000u; // addr[56:32] | type=2
  v8i g1;
  g1[0] = (int)(2u << 16);    // data_size = 4B
  g1[1] = (int)(64u << 16);   // tensor_dim0 = 64
  g1[2] = (int)(1u << 16);    // tensor_dim1 = 1
  g1[3] = (int)(64u << 16);   // tile_dim0 = 64
  g1[4] = 1;                  // tile_dim1 = 1
  g1[5] = 64;                 // tensor_dim0_stride = 64
  g1[6] = 0; g1[7] = 0;
  v4i z4 = {0, 0, 0, 0};
#if defined(__clang_major__) && (__clang_major__ >= 23)
  v8i z8 = {0, 0, 0, 0, 0, 0, 0, 0};
  __builtin_amdgcn_tensor_load_to_lds(g0, g1, z4, z4, z8, 0);
#else
  __builtin_amdgcn_tensor_load_to_lds(g0, g1, z4, z4, 0);
#endif
#if __has_builtin(__builtin_amdgcn_s_wait_tensorcnt)
  __builtin_amdgcn_s_wait_tensorcnt(0);
#endif
  __syncthreads();
  if (threadIdx.x == 0) sink[0] = lbuf[0];
#else
  if (threadIdx.x == 0) sink[0] = src[0];
#endif
}

// ---------------------------------------------------------------- weight pack: OIHW f32 -> [Cout][Kpad] f16 (zero-padded K)
__global__ void pack_w_kernel(const float* __restrict__ w, _Float16* __restrict__ out,
                              int total, int K, int Kpad) {
  int i = blockIdx.x * blockDim.x + threadIdx.x;
  if (i >= total) return;
  int co = i / Kpad, k = i - co * Kpad;
  out[i] = (k < K) ? (_Float16)w[(size_t)co * K + k] : (_Float16)0.f;
}

// ---------------------------------------------------------------- implicit-GEMM conv: C[M][N] = act(alpha*(Wpack * im2col(X) + bias))
// Block tile 64Mx64N, 8 waves; each wave: 1 A-fragment shared by 2 N-fragments -> 2 WMMAs / K-step.
__global__ void __launch_bounds__(256)
gemm_conv_wmma(const _Float16* __restrict__ Aw, int lda,
               const float* __restrict__ Xsrc, int Bn, int Cin, int H, int W,
               int Ho, int Wo, int kh, int kw, int stride, int ph, int pw,
               float* __restrict__ Cc, const float* __restrict__ bias,
               int M, int Nn, int Kpad, float alpha, int act) {
  __shared__ alignas(32) _Float16 ldsB[64 * LDSB];
  int tid = threadIdx.x;
  int lane = tid & 31, wv = tid >> 5;
  int mw = wv >> 1, nw = wv & 1;
  int nblock = blockIdx.x * 64;
  int mblock = blockIdx.y * 64;
  int mfrag = mblock + mw * 16 + (lane & 15);
  if (mfrag >= M) mfrag = M - 1;          // clamp; writes masked later
  int khalf = lane >> 4;
  int K = Cin * kh * kw;
  int khkw = kh * kw, HoWo = Ho * Wo;
  __builtin_prefetch(Aw + (size_t)mfrag * lda, 0, 1);  // global_prefetch_b8
  // staging coords: this thread stages 8 consecutive n for one k per K-step
  int kk_s  = tid >> 3;          // 0..31
  int nn0_s = (tid & 7) * 8;     // 0..56
  v8f acc0 = {}, acc1 = {};
  for (int kc = 0; kc < Kpad; kc += 32) {
    // ---- stage B tile (im2col on the fly), transposed [n][k] into LDS
    int k = kc + kk_s;
    int ci = 0, r = 0, ss = 0;
    bool kvalid = (k < K);
    if (kvalid) {
      ci = k / khkw;  int rem = k - ci * khkw;
      r  = rem / kw;  ss = rem - r * kw;
    }
    const float* srcp = Xsrc + (size_t)ci * Bn * HoWo * 0; // placate compiler (recomputed below)
    (void)srcp;
#pragma unroll
    for (int j = 0; j < 8; ++j) {
      int nn = nn0_s + j;
      int n = nblock + nn;
      float v = 0.f;
      if (kvalid && n < Nn) {
        int b  = n / HoWo;  int r2 = n - b * HoWo;
        int oh = r2 / Wo;   int ow = r2 - oh * Wo;
        int ih = oh * stride - ph + r;
        int iw = ow * stride - pw + ss;
        if (ih >= 0 && ih < H && iw >= 0 && iw < W)
          v = Xsrc[((size_t)ci * Bn + b) * (size_t)(H * W) + (size_t)ih * W + iw];
      }
      ldsB[nn * LDSB + kk_s] = (_Float16)v;
    }
    __syncthreads();
    // ---- A fragment: contiguous k order, same as B (lanes 0-15: K 0..15; 16-31: K 16..31)
    const _Float16* pA = Aw + (size_t)mfrag * lda + kc + khalf * 16;
    union { v16h v; v8h h[2]; } af;
    af.h[0] = *(const v8h*)pA;
    af.h[1] = *(const v8h*)(pA + 8);
    // ---- two B fragments (N and N+16): lanes 0-15 K=0..15, lanes 16-31 K=16..31
    int nl = nw * 32 + (lane & 15);
    v16h bf0 = *(const v16h*)(&ldsB[nl * LDSB + khalf * 16]);
    v16h bf1 = *(const v16h*)(&ldsB[(nl + 16) * LDSB + khalf * 16]);
    acc0 = __builtin_amdgcn_wmma_f32_16x16x32_f16(false, af.v, false, bf0,
                                                  (short)0, acc0, false, false);
    acc1 = __builtin_amdgcn_wmma_f32_16x16x32_f16(false, af.v, false, bf1,
                                                  (short)0, acc1, false, false);
    __syncthreads();
  }
  int ng0 = nblock + nw * 32 + (lane & 15);
#pragma unroll
  for (int r = 0; r < 8; ++r) {
    int mg = mblock + mw * 16 + r + 8 * khalf;
    if (mg < M) {
      float bv = bias ? bias[mg] : 0.f;
      if (ng0 < Nn)
        Cc[(size_t)mg * Nn + ng0] = act_f(alpha * (acc0[r] + bv), act);
      if (ng0 + 16 < Nn)
        Cc[(size_t)mg * Nn + ng0 + 16] = act_f(alpha * (acc1[r] + bv), act);
    }
  }
}

// ---------------------------------------------------------------- plain f16 matrix GEMM (correlation volume), 64x64 tile
__global__ void __launch_bounds__(256)
gemm_mat_wmma(const _Float16* __restrict__ Aw, int lda,
              const _Float16* __restrict__ Bm, int ldb,
              float* __restrict__ Cc, int ldc, const float* __restrict__ bias,
              int M, int Nn, int Kpad, float alpha, int act) {
  __shared__ alignas(32) _Float16 ldsB[64 * LDSB];
  int tid = threadIdx.x;
  int lane = tid & 31, wv = tid >> 5;
  int mw = wv >> 1, nw = wv & 1;
  int nblock = blockIdx.x * 64;
  int mblock = blockIdx.y * 64;
  int mfrag = mblock + mw * 16 + (lane & 15);
  if (mfrag >= M) mfrag = M - 1;
  int khalf = lane >> 4;
  int kk_s  = tid >> 3;
  int nn0_s = (tid & 7) * 8;
  v8f acc0 = {}, acc1 = {};
  for (int kc = 0; kc < Kpad; kc += 32) {
    const _Float16* brow = Bm + (size_t)(kc + kk_s) * ldb;
#pragma unroll
    for (int j = 0; j < 8; ++j) {
      int nn = nn0_s + j;
      int n = nblock + nn;
      _Float16 v = (_Float16)0.f;
      if (n < Nn) v = brow[n];
      ldsB[nn * LDSB + kk_s] = v;
    }
    __syncthreads();
    const _Float16* pA = Aw + (size_t)mfrag * lda + kc + khalf * 16;
    union { v16h v; v8h h[2]; } af;
    af.h[0] = *(const v8h*)pA;
    af.h[1] = *(const v8h*)(pA + 8);
    int nl = nw * 32 + (lane & 15);
    v16h bf0 = *(const v16h*)(&ldsB[nl * LDSB + khalf * 16]);
    v16h bf1 = *(const v16h*)(&ldsB[(nl + 16) * LDSB + khalf * 16]);
    acc0 = __builtin_amdgcn_wmma_f32_16x16x32_f16(false, af.v, false, bf0,
                                                  (short)0, acc0, false, false);
    acc1 = __builtin_amdgcn_wmma_f32_16x16x32_f16(false, af.v, false, bf1,
                                                  (short)0, acc1, false, false);
    __syncthreads();
  }
  int ng0 = nblock + nw * 32 + (lane & 15);
#pragma unroll
  for (int r = 0; r < 8; ++r) {
    int mg = mblock + mw * 16 + r + 8 * khalf;
    if (mg < M) {
      float bv = bias ? bias[mg] : 0.f;
      if (ng0 < Nn)
        Cc[(size_t)mg * ldc + ng0] = act_f(alpha * (acc0[r] + bv), act);
      if (ng0 + 16 < Nn)
        Cc[(size_t)mg * ldc + ng0 + 16] = act_f(alpha * (acc1[r] + bv), act);
    }
  }
}

// ---------------------------------------------------------------- norms (instance: g==null, groups=C*B, seg=HW; batch: groups=C, seg=B*HW)
__global__ void norm_kernel(float* __restrict__ x, const float* __restrict__ g,
                            const float* __restrict__ be, int seg, int relu) {
  __shared__ float s1[256], s2[256];
  float* base = x + (size_t)blockIdx.x * seg;
  int tid = threadIdx.x;
  float a = 0.f, b = 0.f;
  for (int i = tid; i < seg; i += 256) { float v = base[i]; a += v; b += v * v; }
  s1[tid] = a; s2[tid] = b;
  __syncthreads();
  for (int o = 128; o > 0; o >>= 1) {
    if (tid < o) { s1[tid] += s1[tid + o]; s2[tid] += s2[tid + o]; }
    __syncthreads();
  }
  float mean = s1[0] / (float)seg;
  float var  = s2[0] / (float)seg - mean * mean;
  float rs = rsqrtf(var + 1e-5f);
  float gamma = g ? g[blockIdx.x] : 1.f;
  float beta  = be ? be[blockIdx.x] : 0.f;
  for (int i = tid; i < seg; i += 256) {
    float v = (base[i] - mean) * rs * gamma + beta;
    if (relu) v = v > 0.f ? v : 0.f;
    base[i] = v;
  }
}

// ---------------------------------------------------------------- elementwise kernels
__global__ void unary_kernel(float* __restrict__ o, const float* __restrict__ x, int n, int mode) {
  int i = blockIdx.x * blockDim.x + threadIdx.x;
  if (i >= n) return;
  float v = x[i];
  if (mode == 1) v = v > 0.f ? v : 0.f;
  else if (mode == 2) v = 1.f / (1.f + __expf(-v));
  else if (mode == 3) v = tanhf(v);
  else if (mode == 4) v = 2.f * v - 1.f;
  o[i] = v;
}
__global__ void add_relu_kernel(float* __restrict__ o, const float* __restrict__ a,
                                const float* __restrict__ b, int n) {
  int i = blockIdx.x * blockDim.x + threadIdx.x;
  if (i >= n) return;
  float v = a[i] + b[i];
  o[i] = v > 0.f ? v : 0.f;
}
__global__ void cvt16_kernel(_Float16* __restrict__ o, const float* __restrict__ x, int n) {
  int i = blockIdx.x * blockDim.x + threadIdx.x;
  if (i < n) o[i] = (_Float16)x[i];
}
// f1 [C][B*HW] f32 -> f1t [B][HW][C] f16
__global__ void transpose_kernel(_Float16* __restrict__ dst, const float* __restrict__ src,
                                 int C, int Bn, int HW) {
  int i = blockIdx.x * blockDim.x + threadIdx.x;
  int tot = Bn * HW * C;
  if (i >= tot) return;
  int k = i % C;
  int m = (i / C) % HW;
  int b = i / (C * HW);
  dst[i] = (_Float16)src[(size_t)k * (Bn * HW) + (size_t)b * HW + m];
}
__global__ void avgpool_kernel(float* __restrict__ out, const float* __restrict__ in,
                               int N, int hi, int wi) {
  int ho = hi >> 1, wo = wi >> 1;
  int tot = N * ho * wo;
  int i = blockIdx.x * blockDim.x + threadIdx.x;
  if (i >= tot) return;
  int n = i / (ho * wo);
  int r = i - n * (ho * wo);
  int y = r / wo, x = r - y * wo;
  const float* p = in + ((size_t)n * hi + 2 * y) * wi + 2 * x;
  out[i] = 0.25f * (p[0] + p[1] + p[wi] + p[wi + 1]);
}
__global__ void coords_init_kernel(float* cx, float* cy, int n) {
  int i = blockIdx.x * blockDim.x + threadIdx.x;
  if (i >= n) return;
  int p = i & 1023;
  cx[i] = (float)(p & 31);
  cy[i] = (float)(p >> 5);
}
__global__ void flow_kernel(float* fx, float* fy, const float* cx, const float* cy, int n) {
  int i = blockIdx.x * blockDim.x + threadIdx.x;
  if (i >= n) return;
  int p = i & 1023;
  fx[i] = cx[i] - (float)(p & 31);
  fy[i] = cy[i] - (float)(p >> 5);
}
__global__ void coords_update_kernel(float* cx, float* cy, const float* delta, int n) {
  int i = blockIdx.x * blockDim.x + threadIdx.x;
  if (i >= n) return;
  cx[i] += delta[i];
  cy[i] += delta[n + i];
}
__global__ void rh_mul_kernel(float* __restrict__ rh, const float* __restrict__ r,
                              const float* __restrict__ h, int n) {
  int i = blockIdx.x * blockDim.x + threadIdx.x;
  if (i < n) rh[i] = r[i] * h[i];
}
__global__ void gru_comb_kernel(float* __restrict__ h, const float* __restrict__ z,
                                const float* __restrict__ q, int n) {
  int i = blockIdx.x * blockDim.x + threadIdx.x;
  if (i < n) h[i] = (1.f - z[i]) * h[i] + z[i] * q[i];
}

// ---------------------------------------------------------------- correlation lookup (4-level pyramid, 9x9 window, bilinear, zero-pad)
__global__ void lookup_kernel(const float* __restrict__ p0, const float* __restrict__ p1,
                              const float* __restrict__ p2, const float* __restrict__ p3,
                              const float* __restrict__ cx, const float* __restrict__ cy,
                              float* __restrict__ out, int BHW) {
  int idx = blockIdx.x * blockDim.x + threadIdx.x;
  if (idx >= BHW * 81) return;
  int pos = idx / 81, kk = idx - pos * 81;
  float dxv = (float)(kk % 9) - 4.f;
  float dyv = (float)(kk / 9) - 4.f;
  const float* imgs[4] = {p0, p1, p2, p3};
  float cxv = cx[pos], cyv = cy[pos];
#pragma unroll
  for (int l = 0; l < 4; ++l) {
    int wl = 32 >> l, hl = 32 >> l;
    float xs = cxv / (float)(1 << l) + dxv;
    float ys = cyv / (float)(1 << l) + dyv;
    float x0 = floorf(xs), y0 = floorf(ys);
    float wx = xs - x0, wy = ys - y0;
    const float* img = imgs[l] + (size_t)pos * hl * wl;
    auto gat = [&](float xf, float yf) -> float {
      bool valid = (xf >= 0.f) && (xf <= (float)(wl - 1)) &&
                   (yf >= 0.f) && (yf <= (float)(hl - 1));
      int xi = (int)fminf(fmaxf(xf, 0.f), (float)(wl - 1));
      int yi = (int)fminf(fmaxf(yf, 0.f), (float)(hl - 1));
      float v = img[yi * wl + xi];
      return valid ? v : 0.f;
    };
    float v = gat(x0, y0) * (1.f - wx) * (1.f - wy)
            + gat(x0 + 1.f, y0) * wx * (1.f - wy)
            + gat(x0, y0 + 1.f) * (1.f - wx) * wy
            + gat(x0 + 1.f, y0 + 1.f) * wx * wy;
    out[(size_t)(l * 81 + kk) * BHW + pos] = v;
  }
}

// ---------------------------------------------------------------- convex upsample -> NHWC (B,256,256,2)
__global__ void upsample_kernel(float* __restrict__ out, const float* __restrict__ mask,
                                const float* __restrict__ cx, const float* __restrict__ cy) {
  int idx = blockIdx.x * blockDim.x + threadIdx.x;
  if (idx >= BN * 1024 * 64) return;
  int sub = idx & 63;
  int j8 = sub & 7, i8 = sub >> 3;
  int pos = idx >> 6;
  int b = pos >> 10;
  int py = (pos >> 5) & 31, px = pos & 31;
  float mv[9], mx = -1e30f;
#pragma unroll
  for (int p = 0; p < 9; ++p) {
    mv[p] = mask[(size_t)(p * 64 + i8 * 8 + j8) * (BN * 1024) + pos];
    mx = fmaxf(mx, mv[p]);
  }
  float sum = 0.f;
#pragma unroll
  for (int p = 0; p < 9; ++p) { mv[p] = __expf(mv[p] - mx); sum += mv[p]; }
  float fx = 0.f, fy = 0.f;
#pragma unroll
  for (int p = 0; p < 9; ++p) {
    int dy = p / 3 - 1, dx = p % 3 - 1;
    int ny = py + dy, nx = px + dx;
    float w = mv[p] / sum;
    if (ny >= 0 && ny < 32 && nx >= 0 && nx < 32) {
      int np = b * 1024 + ny * 32 + nx;
      fx += w * 8.f * (cx[np] - (float)nx);
      fy += w * 8.f * (cy[np] - (float)ny);
    }
  }
  int Y = py * 8 + i8, X = px * 8 + j8;
  size_t o = (((size_t)b * 256 + Y) * 256 + X) * 2;
  out[o + 0] = fx;
  out[o + 1] = fy;
}

// ================================================================ host side
struct Bump {
  char* base; size_t off;
  void* get(size_t n) { size_t a = (off + 255) & ~(size_t)255; off = a + n; return base + a; }
  float* f32(size_t n) { return (float*)get(n * sizeof(float)); }
  _Float16* f16(size_t n) { return (_Float16*)get(n * sizeof(_Float16)); }
};
struct PCur {
  void* const* din; int i;
  const float* nx() { return (const float*)din[i++]; }
};
static inline int outdim(int H, int k, int st) { return (H + 2 * (k / 2) - k) / st + 1; }

static void conv_run(hipStream_t s, _Float16* wpack, const float* x,
                     int cin, int H, int W, int st, const float* w, const float* b,
                     int cout, int kh, int kw, float* y, int act, float alpha) {
  int K = cin * kh * kw;
  int Kpad = ((K + 31) / 32) * 32;
  int Ho = outdim(H, kh, st), Wo = outdim(W, kw, st);
  int tot = cout * Kpad;
  pack_w_kernel<<<CDIV(tot, 256), 256, 0, s>>>(w, wpack, tot, K, Kpad);
  int Nn = BN * Ho * Wo;
  dim3 g(CDIV(Nn, 64), CDIV(cout, 64));
  gemm_conv_wmma<<<g, 256, 0, s>>>(wpack, Kpad, x, BN, cin, H, W, Ho, Wo,
                                   kh, kw, st, kh / 2, kw / 2,
                                   y, b, cout, Nn, Kpad, alpha, act);
}
static void norm_run(hipStream_t s, bool bn, float* x, int C, int HoWo,
                     const float* g, const float* be, int relu) {
  if (bn) norm_kernel<<<C, 256, 0, s>>>(x, g, be, BN * HoWo, relu);
  else    norm_kernel<<<C * BN, 256, 0, s>>>(x, nullptr, nullptr, HoWo, relu);
}

static const float* res_block(hipStream_t s, PCur& prm, bool bn, const float* x,
                              int cin, int cout, int H, int W, int st,
                              float* y1, float* y2, float* xr, _Float16* wpack) {
  const float* c1w = prm.nx(); const float* c1b = prm.nx();
  const float* c2w = prm.nx(); const float* c2b = prm.nx();
  const float *g1 = nullptr, *be1 = nullptr, *g2 = nullptr, *be2 = nullptr;
  if (bn) { g1 = prm.nx(); be1 = prm.nx(); g2 = prm.nx(); be2 = prm.nx(); }
  bool down = (cin != cout);
  const float *dw = nullptr, *db = nullptr, *dg = nullptr, *dbe = nullptr;
  if (down) { dw = prm.nx(); db = prm.nx(); if (bn) { dg = prm.nx(); dbe = prm.nx(); } }
  int Ho = outdim(H, 3, st), Wo = outdim(W, 3, st);
  conv_run(s, wpack, x, cin, H, W, st, c1w, c1b, cout, 3, 3, y1, 0, 1.f);
  norm_run(s, bn, y1, cout, Ho * Wo, g1, be1, 1);
  conv_run(s, wpack, y1, cout, Ho, Wo, 1, c2w, c2b, cout, 3, 3, y2, 0, 1.f);
  norm_run(s, bn, y2, cout, Ho * Wo, g2, be2, 1);
  const float* res = x;
  if (down) {
    conv_run(s, wpack, x, cin, H, W, st, dw, db, cout, 1, 1, xr, 0, 1.f);
    norm_run(s, bn, xr, cout, Ho * Wo, dg, dbe, 0);
    res = xr;
  }
  int n = cout * BN * Ho * Wo;
  add_relu_kernel<<<CDIV(n, 256), 256, 0, s>>>(y1, res, y2, n);
  return y1;
}

static void run_encoder(hipStream_t s, PCur& prm, bool bn, const float* img,
                        float* out256, float* bA, float* bB, float* bC, float* bD,
                        _Float16* wpack) {
  const float* w = prm.nx(); const float* b = prm.nx();
  const float *g = nullptr, *be = nullptr;
  if (bn) { g = prm.nx(); be = prm.nx(); }
  conv_run(s, wpack, img, 1, 256, 256, 2, w, b, 64, 7, 7, bA, 0, 1.f);
  norm_run(s, bn, bA, 64, 128 * 128, g, be, 1);
  float* pool[4] = {bA, bB, bC, bD};
  const float* x = bA;
  struct Cfg { int cin, cout, H, W, st; };
  Cfg cfg[6] = {{64, 64, 128, 128, 1}, {64, 64, 128, 128, 1},
                {64, 96, 128, 128, 2}, {96, 96, 64, 64, 1},
                {96, 128, 64, 64, 2}, {128, 128, 32, 32, 1}};
  for (int bi = 0; bi < 6; ++bi) {
    float* t[3]; int j = 0;
    for (int k = 0; k < 4; ++k) if (pool[k] != x) t[j++] = pool[k];
    x = res_block(s, prm, bn, x, cfg[bi].cin, cfg[bi].cout,
                  cfg[bi].H, cfg[bi].W, cfg[bi].st, t[0], t[1], t[2], wpack);
  }
  w = prm.nx(); b = prm.nx();
  conv_run(s, wpack, x, 128, 32, 32, 1, w, b, 256, 1, 1, out256, 0, 1.f);
}

extern "C" void kernel_launch(void* const* d_in, const int* in_sizes, int n_in,
                              void* d_out, int out_size, void* d_ws, size_t ws_size,
                              hipStream_t stream) {
  (void)in_sizes; (void)n_in; (void)out_size; (void)ws_size;
  hipStream_t s = stream;
  const float* image_moving = (const float*)d_in[0];
  const float* image_fixed  = (const float*)d_in[1];
  float* out = (float*)d_out;

  const int HW = 1024, N32 = BN * HW;            // latent 32x32 grid
  Bump ws{(char*)d_ws, 0};
  float* img1 = ws.f32((size_t)BN * 65536);
  float* img2 = ws.f32((size_t)BN * 65536);
  size_t poolN = (size_t)64 * BN * 16384;        // 64ch @ 128x128 (max)
  float* bA = ws.f32(poolN); float* bB = ws.f32(poolN);
  float* bC = ws.f32(poolN); float* bD = ws.f32(poolN);
  float* f1  = ws.f32((size_t)256 * N32);
  float* f2  = ws.f32((size_t)256 * N32);
  float* ctx = ws.f32((size_t)256 * N32);
  _Float16* f1t = ws.f16((size_t)BN * HW * 256);
  _Float16* f2h = ws.f16((size_t)256 * N32);
  float* pyr0 = ws.f32((size_t)N32 * 1024);
  float* pyr1 = ws.f32((size_t)N32 * 256);
  float* pyr2 = ws.f32((size_t)N32 * 64);
  float* pyr3 = ws.f32((size_t)N32 * 16);
  float* X     = ws.f32((size_t)384 * N32);  // [0:128]=h [128:256]=ctx [256:382]=m [382:384]=flow
  float* RH    = ws.f32((size_t)384 * N32);
  float* corrF = ws.f32((size_t)324 * N32);
  float* c1o   = ws.f32((size_t)256 * N32);
  float* CF    = ws.f32((size_t)256 * N32);  // [0:192]=me.c2 out, [192:256]=me.f2 out
  float* zb = ws.f32((size_t)128 * N32);
  float* rb = ws.f32((size_t)128 * N32);
  float* qb = ws.f32((size_t)128 * N32);
  float* t256  = ws.f32((size_t)256 * N32);
  float* maskb = ws.f32((size_t)576 * N32);
  float* delta = ws.f32((size_t)2 * N32);
  float* cx = ws.f32(N32);
  float* cy = ws.f32(N32);
  _Float16* wpack = ws.f16((size_t)1 << 20);
  float* sink = ws.f32(64);

  PCur prm{d_in, 3};

  // TDM prefetch of first weight tensor (async-tensor path)
  tdm_prefetch_kernel<<<1, 32, 0, s>>>((const float*)d_in[3], sink);

  int nimg = BN * 65536;
  unary_kernel<<<CDIV(nimg, 256), 256, 0, s>>>(img1, image_fixed, nimg, 4);   // 2x-1
  unary_kernel<<<CDIV(nimg, 256), 256, 0, s>>>(img2, image_moving, nimg, 4);

  // feature encoders (shared 'f' params) + context encoder ('c', batchnorm)
  int fstart = prm.i;
  run_encoder(s, prm, false, img1, f1, bA, bB, bC, bD, wpack);
  int fend = prm.i;
  prm.i = fstart;
  run_encoder(s, prm, false, img2, f2, bA, bB, bC, bD, wpack);
  prm.i = fend;
  run_encoder(s, prm, true, img1, ctx, bA, bB, bC, bD, wpack);

  // hidden = tanh(ctx[:128]); context = relu(ctx[128:])
  unary_kernel<<<CDIV(128 * N32, 256), 256, 0, s>>>(X, ctx, 128 * N32, 3);
  unary_kernel<<<CDIV(128 * N32, 256), 256, 0, s>>>(X + (size_t)128 * N32,
                                                    ctx + (size_t)128 * N32, 128 * N32, 1);

  // correlation volume via WMMA GEMM: corr[b] = f1[b]^T * f2[b] / 16
  cvt16_kernel<<<CDIV(256 * N32, 256), 256, 0, s>>>(f2h, f2, 256 * N32);
  transpose_kernel<<<CDIV(BN * HW * 256, 256), 256, 0, s>>>(f1t, f1, 256, BN, HW);
  for (int b = 0; b < BN; ++b) {
    dim3 g(1024 / 64, 1024 / 64);
    gemm_mat_wmma<<<g, 256, 0, s>>>(f1t + (size_t)b * HW * 256, 256,
                                    f2h + (size_t)b * HW, N32,
                                    pyr0 + (size_t)b * HW * 1024, 1024, nullptr,
                                    1024, 1024, 256, 0.0625f, 0);
  }
  avgpool_kernel<<<CDIV(N32 * 256, 256), 256, 0, s>>>(pyr1, pyr0, N32, 32, 32);
  avgpool_kernel<<<CDIV(N32 * 64, 256), 256, 0, s>>>(pyr2, pyr1, N32, 16, 16);
  avgpool_kernel<<<CDIV(N32 * 16, 256), 256, 0, s>>>(pyr3, pyr2, N32, 8, 8);

  coords_init_kernel<<<CDIV(N32, 256), 256, 0, s>>>(cx, cy, N32);

  float* flowp = X + (size_t)382 * N32;
  int itStart = prm.i;
  for (int it = 0; it < 12; ++it) {
    prm.i = itStart;
    lookup_kernel<<<CDIV(N32 * 81, 256), 256, 0, s>>>(pyr0, pyr1, pyr2, pyr3,
                                                      cx, cy, corrF, N32);
    flow_kernel<<<CDIV(N32, 256), 256, 0, s>>>(flowp, flowp + N32, cx, cy, N32);
    const float *w, *b;
    // motion encoder
    w = prm.nx(); b = prm.nx();
    conv_run(s, wpack, corrF, 324, 32, 32, 1, w, b, 256, 1, 1, c1o, 1, 1.f);
    w = prm.nx(); b = prm.nx();
    conv_run(s, wpack, c1o, 256, 32, 32, 1, w, b, 192, 3, 3, CF, 1, 1.f);
    w = prm.nx(); b = prm.nx();
    conv_run(s, wpack, flowp, 2, 32, 32, 1, w, b, 128, 7, 7, c1o, 1, 1.f);
    w = prm.nx(); b = prm.nx();
    conv_run(s, wpack, c1o, 128, 32, 32, 1, w, b, 64, 3, 3, CF + (size_t)192 * N32, 1, 1.f);
    w = prm.nx(); b = prm.nx();
    conv_run(s, wpack, CF, 256, 32, 32, 1, w, b, 126, 3, 3, X + (size_t)256 * N32, 1, 1.f);
    // x part of RH (= context|motion|flow), reused by both GRU passes
    hipMemcpyAsync(RH + (size_t)128 * N32, X + (size_t)128 * N32,
                   (size_t)256 * N32 * sizeof(float), hipMemcpyDeviceToDevice, s);
    // horizontal GRU (1x5)
    w = prm.nx(); b = prm.nx();
    conv_run(s, wpack, X, 384, 32, 32, 1, w, b, 128, 1, 5, zb, 2, 1.f);
    w = prm.nx(); b = prm.nx();
    conv_run(s, wpack, X, 384, 32, 32, 1, w, b, 128, 1, 5, rb, 2, 1.f);
    w = prm.nx(); b = prm.nx();
    rh_mul_kernel<<<CDIV(128 * N32, 256), 256, 0, s>>>(RH, rb, X, 128 * N32);
    conv_run(s, wpack, RH, 384, 32, 32, 1, w, b, 128, 1, 5, qb, 3, 1.f);
    gru_comb_kernel<<<CDIV(128 * N32, 256), 256, 0, s>>>(X, zb, qb, 128 * N32);
    // vertical GRU (5x1)
    w = prm.nx(); b = prm.nx();
    conv_run(s, wpack, X, 384, 32, 32, 1, w, b, 128, 5, 1, zb, 2, 1.f);
    w = prm.nx(); b = prm.nx();
    conv_run(s, wpack, X, 384, 32, 32, 1, w, b, 128, 5, 1, rb, 2, 1.f);
    w = prm.nx(); b = prm.nx();
    rh_mul_kernel<<<CDIV(128 * N32, 256), 256, 0, s>>>(RH, rb, X, 128 * N32);
    conv_run(s, wpack, RH, 384, 32, 32, 1, w, b, 128, 5, 1, qb, 3, 1.f);
    gru_comb_kernel<<<CDIV(128 * N32, 256), 256, 0, s>>>(X, zb, qb, 128 * N32);
    // flow head
    w = prm.nx(); b = prm.nx();
    conv_run(s, wpack, X, 128, 32, 32, 1, w, b, 256, 3, 3, t256, 1, 1.f);
    w = prm.nx(); b = prm.nx();
    conv_run(s, wpack, t256, 256, 32, 32, 1, w, b, 2, 3, 3, delta, 0, 1.f);
    coords_update_kernel<<<CDIV(N32, 256), 256, 0, s>>>(cx, cy, delta, N32);
    // upsample (last_only=1 -> final iter only)
    if (it == 11) {
      w = prm.nx(); b = prm.nx();
      conv_run(s, wpack, X, 128, 32, 32, 1, w, b, 256, 3, 3, t256, 1, 1.f);
      w = prm.nx(); b = prm.nx();
      conv_run(s, wpack, t256, 256, 32, 32, 1, w, b, 576, 1, 1, maskb, 0, 0.25f);
      upsample_kernel<<<CDIV(BN * 1024 * 64, 256), 256, 0, s>>>(out, maskb, cx, cy);
    }
  }
}
